// DeepGate3_39067022524840
// MI455X (gfx1250) — compile-verified, matrix operand order checked by reading, not run on hardware
//
#include <hip/hip_runtime.h>
#include <hip/hip_bf16.h>

// ---------------------------------------------------------------------------
// Problem constants (from reference)
// ---------------------------------------------------------------------------
#define NN   200000
#define HH   10000
#define DD   128
#define FFH  2048
#define MLPH 256
#define TTN  64

typedef __attribute__((ext_vector_type(16))) _Float16 v16h;
typedef __attribute__((ext_vector_type(8)))  float    v8f;
typedef unsigned int u32x4 __attribute__((ext_vector_type(4)));
typedef int          i32x4 __attribute__((ext_vector_type(4)));
typedef int          i32x8 __attribute__((ext_vector_type(8)));

// ---------------------------------------------------------------------------
// WMMA helpers (CDNA5 wave32, 16x16x32 f16 -> f32)
// ---------------------------------------------------------------------------
__device__ inline v8f wmma16(v16h a, v16h b, v8f c) {
  return __builtin_amdgcn_wmma_f32_16x16x32_f16(false, a, false, b, (short)0, c,
                                                false, false);
}

// Load a 16x32 f16 fragment from a row-major tile (row stride `ld` elements).
// Works for A (rows = M) and for B given in BT = [N][K] layout (rows = N).
__device__ inline v16h load_frag_rm(const _Float16* p0, int ld) {
  const int lane = threadIdx.x & 31;
  const int r    = lane & 15;
  const int kb   = (lane & 16) ? 8 : 0;
  const _Float16* p = p0 + (size_t)r * ld + kb;
  union { uint4 q[2]; v16h v; } f;
  f.q[0] = *(const uint4*)(p);
  f.q[1] = *(const uint4*)(p + 16);
  return f.v;
}

// ---------------------------------------------------------------------------
// TDM: stage `nelem` contiguous f32 from global into LDS at byte offset
// lds_off. One issue per calling wave; waits TENSORcnt itself; caller adds the barrier.
// Descriptor per CDNA5 ISA ch.8 (groups 0/1; groups 2/3 zero).
// ---------------------------------------------------------------------------
__device__ inline void tdm_load_f32_1d(unsigned lds_off, const void* gptr,
                                       unsigned nelem) {
  unsigned long long ga = (unsigned long long)(uintptr_t)gptr;
  u32x4 g0 = { 1u,                                   // count=1, user mode
               lds_off,                               // lds_addr
               (unsigned)(ga & 0xFFFFFFFFu),          // global_addr[31:0]
               (unsigned)((ga >> 32) & 0x01FFFFFFu)   // global_addr[56:32]
                 | 0x80000000u };                     // type=2 (bits 127:126)
  i32x8 g1 = { (int)(2u << 16),                       // data_size = 4B
               (int)((nelem & 0xFFFFu) << 16),        // tensor_dim0[15:0]
               (int)((nelem >> 16) | (1u << 16)),     // tensor_dim0[31:16] | tensor_dim1=1
               (int)(nelem << 16),                    // tile_dim0
               1,                                     // tile_dim1=1
               (int)nelem,                            // tensor_dim0_stride
               0, 0 };
  i32x4 z4 = { 0, 0, 0, 0 };
#if __clang_major__ >= 23
  i32x8 z8 = { 0, 0, 0, 0, 0, 0, 0, 0 };
  __builtin_amdgcn_tensor_load_to_lds(g0, g1, z4, z4, z8, 0);
#else
  __builtin_amdgcn_tensor_load_to_lds(g0, g1, z4, z4, 0);
#endif
  __builtin_amdgcn_s_wait_tensorcnt(0);
}

// ---------------------------------------------------------------------------
// Tiled GEMM: C[M][N] = A[M][K](f16) * BT[N][K](f16)^T + bias, f32 out.
// One wave computes a 16x64 strip (4 WMMA tiles) so each A fragment is
// reused 4x -> 4x less A traffic (A stream dominates HBM BW here).
// Block = 256 threads = 8 waves. Requires N % 64 == 0, M % 16 == 0.
// ---------------------------------------------------------------------------
__global__ __launch_bounds__(256)
void gemm16x4_kernel(const _Float16* __restrict__ A, const _Float16* __restrict__ BT,
                     const float* __restrict__ bias, float* __restrict__ C,
                     int M, int N, int K)
{
  const int wave = blockIdx.x * 8 + (threadIdx.x >> 5);
  const int ngrp = N >> 6;
  const int total = (M >> 4) * ngrp;
  if (wave >= total) return;
  const int mt = wave / ngrp, ng = wave % ngrp;
  const _Float16* a0 = A  + (size_t)mt * 16 * K;
  const _Float16* b0 = BT + (size_t)ng * 64 * K;
  v8f acc0 = {}, acc1 = {}, acc2 = {}, acc3 = {};
  for (int k = 0; k < K; k += 32) {
    v16h a = load_frag_rm(a0 + k, K);
    acc0 = wmma16(a, load_frag_rm(b0 + k,              K), acc0);
    acc1 = wmma16(a, load_frag_rm(b0 + (size_t)16 * K + k, K), acc1);
    acc2 = wmma16(a, load_frag_rm(b0 + (size_t)32 * K + k, K), acc2);
    acc3 = wmma16(a, load_frag_rm(b0 + (size_t)48 * K + k, K), acc3);
  }
  const int lane = threadIdx.x & 31;
  const int nloc = lane & 15;
  const int mb   = mt * 16 + ((lane & 16) ? 8 : 0);
  v8f accs[4] = { acc0, acc1, acc2, acc3 };
#pragma unroll
  for (int t = 0; t < 4; ++t) {
    int n = ng * 64 + t * 16 + nloc;
    float bv = bias ? bias[n] : 0.0f;
#pragma unroll
    for (int r = 0; r < 8; ++r)
      C[(size_t)(mb + r) * N + n] = accs[t][r] + bv;
  }
}

// ---------------------------------------------------------------------------
// Row LayerNorm (width == blockDim == 256) + ReLU, f32 in -> f16 out (mlp3)
// ---------------------------------------------------------------------------
__global__ __launch_bounds__(256)
void ln_relu_kernel(const float* __restrict__ X, const float* __restrict__ g,
                    const float* __restrict__ b, _Float16* __restrict__ Y)
{
  const int tid = threadIdx.x;
  const float* x = X + (size_t)blockIdx.x * 256;
  __shared__ float red[256];
  float v = x[tid];
  red[tid] = v; __syncthreads();
  for (int s = 128; s > 0; s >>= 1) { if (tid < s) red[tid] += red[tid + s]; __syncthreads(); }
  float mean = red[0] * (1.0f / 256.0f);
  __syncthreads();
  float d = v - mean;
  red[tid] = d * d; __syncthreads();
  for (int s = 128; s > 0; s >>= 1) { if (tid < s) red[tid] += red[tid + s]; __syncthreads(); }
  float var = red[0] * (1.0f / 256.0f);
  float y = d * rsqrtf(var + 1e-5f) * g[tid] + b[tid];
  Y[(size_t)blockIdx.x * 256 + tid] = (_Float16)fmaxf(y, 0.0f);
}

// prob = H(f16)[M][256] . w2(f32)[256] + b2 ; one wave per row
__global__ __launch_bounds__(256)
void prob_kernel(const _Float16* __restrict__ Hh, const float* __restrict__ w2,
                 const float* __restrict__ b2, float* __restrict__ prob, int M)
{
  const int row = blockIdx.x * 8 + (threadIdx.x >> 5);
  if (row >= M) return;
  const int lane = threadIdx.x & 31;
  const _Float16* h = Hh + (size_t)row * 256;
  float s = 0.0f;
  for (int k = lane; k < 256; k += 32) s += (float)h[k] * w2[k];
  for (int off = 16; off; off >>= 1) s += __shfl_xor(s, off, 32);
  if (lane == 0) prob[row] = s + b2[0];
}

// ---------------------------------------------------------------------------
// f32 -> f16 conversion and transpose-conversion (weight prep)
// ---------------------------------------------------------------------------
__global__ __launch_bounds__(256)
void conv_f16_kernel(const float* __restrict__ in, _Float16* __restrict__ out, long n)
{
  long i = (long)blockIdx.x * 256 + threadIdx.x;
  if (i < n) out[i] = (_Float16)in[i];
}
// in[R][C] (f32) -> out[C][R] (f16)
__global__ __launch_bounds__(256)
void tconv_f16_kernel(const float* __restrict__ in, _Float16* __restrict__ out,
                      int R, int C)
{
  int i = blockIdx.x * 256 + threadIdx.x;
  if (i >= R * C) return;
  int r = i / C, c = i % C;
  out[(size_t)c * R + r] = (_Float16)in[i];
}

// ---------------------------------------------------------------------------
// Build encoder input sequences: x[h] = 16 rows x 128 (rows 0..9 real, 10..15 pad)
// ---------------------------------------------------------------------------
__global__ __launch_bounds__(128)
void build_seq_kernel(const float* __restrict__ hf, const int* __restrict__ hop_pi,
                      const int* __restrict__ stats, const int* __restrict__ hop_po,
                      const float* __restrict__ cls, const float* __restrict__ dc,
                      const float* __restrict__ zero, const float* __restrict__ one,
                      const float* __restrict__ pos, float* __restrict__ xbuf,
                      int* __restrict__ kvmask)
{
  const int h = blockIdx.x;
  const int d = threadIdx.x;                  // 0..127
  int st[8], pi[8], ord[8];
  for (int i = 0; i < 8; ++i) { st[i] = stats[h * 8 + i]; pi[i] = hop_pi[h * 8 + i]; }
  int k = 0;
  for (int i = 0; i < 8; ++i) if (st[i] != -1) ord[k++] = i;
  int kk = k;
  for (int i = 0; i < 8; ++i) if (st[i] == -1) ord[kk++] = i;
  int dpad = 6 - k; if (dpad < 0) dpad = 0;

  float* xr = xbuf + (size_t)h * 16 * 128;
  xr[d] = cls[d] + pos[d];                    // token 0: CLS
  unsigned mask = 1u | (1u << 9);
  for (int t = 1; t <= 8; ++t) {
    int tt = t - 1;
    bool isdc = tt < dpad;
    int src = tt - dpad;
    bool incomp = (src >= 0) && (src < k);
    float val;
    if (isdc) val = dc[d];
    else if (incomp) {
      int slot = ord[src];
      int s2 = st[slot];
      if (s2 == 0)      val = zero[d];
      else if (s2 == 1) val = one[d];
      else              val = hf[(size_t)pi[slot] * 128 + d];
    } else val = 0.0f;
    xr[t * 128 + d] = val + pos[t * 128 + d];
    if (isdc || incomp) mask |= (1u << t);
  }
  xr[9 * 128 + d] = hf[(size_t)hop_po[h] * 128 + d] + pos[9 * 128 + d];
  for (int t = 10; t < 16; ++t) xr[t * 128 + d] = 0.0f;
  if (d == 0) kvmask[h] = (int)mask;
}

// ---------------------------------------------------------------------------
// 16-row LayerNorm helper (W = 128 or 512), wave-per-row, 8 waves / 16 rows
// ---------------------------------------------------------------------------
template<int W>
__device__ inline void ln_rows16(const float* in, const float* g, const float* b,
                                 float* outf, _Float16* outh)
{
  const int wave = threadIdx.x >> 5, lane = threadIdx.x & 31;
  constexpr int PE = W / 32;
  for (int r = wave; r < 16; r += 8) {
    const float* x = in + (size_t)r * W;
    float v[PE]; float s = 0.0f;
#pragma unroll
    for (int i = 0; i < PE; ++i) { v[i] = x[lane + i * 32]; s += v[i]; }
    for (int off = 16; off; off >>= 1) s += __shfl_xor(s, off, 32);
    float mean = s * (1.0f / (float)W);
    float q = 0.0f;
#pragma unroll
    for (int i = 0; i < PE; ++i) { v[i] -= mean; q += v[i] * v[i]; }
    for (int off = 16; off; off >>= 1) q += __shfl_xor(q, off, 32);
    float inv = rsqrtf(q * (1.0f / (float)W) + 1e-5f);
#pragma unroll
    for (int i = 0; i < PE; ++i) {
      int c = lane + i * 32;
      float y = v[i] * inv * g[c] + b[c];
      if (outf) outf[(size_t)r * W + c] = y;
      if (outh) outh[(size_t)r * W + c] = (_Float16)y;
    }
  }
}

// ---------------------------------------------------------------------------
// One full encoder layer for one hop per block (256 threads = 8 waves).
// x tile staged by the Tensor Data Mover (async DMA, TENSORcnt); QKV / Wo /
// FF GEMMs via WMMA; attention + LN in f32 VALU; weight prefetch via
// global_prefetch_b8.
// Dynamic LDS: [0,8K) = xs (16x128 f32), [8K,72K) = hb (16x2048 f16).
// ---------------------------------------------------------------------------
__global__ __launch_bounds__(256)
void enc_layer_kernel(float* __restrict__ xbuf, const int* __restrict__ kvmask,
                      const _Float16* __restrict__ Wqkv, const float* __restrict__ bqkv,
                      const _Float16* __restrict__ Wo,   const float* __restrict__ bo,
                      const float* __restrict__ g1, const float* __restrict__ b1,
                      const _Float16* __restrict__ W1,   const float* __restrict__ bf1,
                      const _Float16* __restrict__ W2,   const float* __restrict__ bf2,
                      const float* __restrict__ g2, const float* __restrict__ b2)
{
  __shared__ __align__(16) _Float16 xh [16 * 128];   // x as f16 (A operand)
  __shared__ __align__(16) float    qkv[16 * 384];   // q|k|v
  __shared__ __align__(16) float    sc [4 * 16 * 16];// scores -> attn weights
  __shared__ __align__(16) float    oo [16 * 128];   // attn out / pre-LN buf
  __shared__ __align__(16) _Float16 oh [16 * 128];   // attn out f16
  extern __shared__ __align__(16) char dynlds[];
  float*    xs = (float*)dynlds;                     // 16x128 f32 (TDM target)
  _Float16* hb = (_Float16*)(dynlds + 16 * 128 * 4); // 16x2048 FF hidden (f16)

  const int h    = blockIdx.x;
  const int wave = threadIdx.x >> 5;
  const int lane = threadIdx.x & 31;
  float* xg = xbuf + (size_t)h * 16 * 128;
  const unsigned mask = (unsigned)kvmask[h];

  // 1) DMA x tile into LDS via TDM (wave 0 issues; barrier publishes)
  if (wave == 0)
    tdm_load_f32_1d(__builtin_amdgcn_groupstaticsize(), xg, 16 * 128);
  __syncthreads();
  for (int i = threadIdx.x; i < 16 * 128; i += 256) xh[i] = (_Float16)xs[i];
  __syncthreads();

  // 2) QKV: N=384 -> 24 tiles, 3 per wave
  for (int nt = wave; nt < 24; nt += 8) {
    if (nt + 8 < 24) __builtin_prefetch(Wqkv + (size_t)(nt + 8) * 16 * 128, 0, 3);
    v8f acc = {};
    const _Float16* b0p = Wqkv + (size_t)nt * 16 * 128;
    for (int k = 0; k < 128; k += 32)
      acc = wmma16(load_frag_rm(xh + k, 128), load_frag_rm(b0p + k, 128), acc);
    int n = nt * 16 + (lane & 15); int mb = (lane & 16) ? 8 : 0;
    float bv = bqkv[n];
#pragma unroll
    for (int r = 0; r < 8; ++r) qkv[(mb + r) * 384 + n] = acc[r] + bv;
  }
  __syncthreads();

  // 3) attention scores (4 heads x 16 q x 16 k), masked, scaled
  for (int e = threadIdx.x; e < 4 * 16 * 16; e += 256) {
    int hd = e >> 8, q = (e >> 4) & 15, kk = e & 15;
    const float* qp = qkv + q  * 384 + hd * 32;
    const float* kp = qkv + kk * 384 + 128 + hd * 32;
    float s = 0.0f;
    for (int d = 0; d < 32; ++d) s += qp[d] * kp[d];
    s *= 0.17677669529663687f;                 // 1/sqrt(32)
    if (!((mask >> kk) & 1u)) s = -1e9f;       // padded/invalid keys masked
    sc[e] = s;
  }
  __syncthreads();
  // softmax per (head, q)
  if (threadIdx.x < 64) {
    float* row = sc + threadIdx.x * 16;
    float m = row[0];
    for (int i = 1; i < 16; ++i) m = fmaxf(m, row[i]);
    float ssum = 0.0f;
    for (int i = 0; i < 16; ++i) { float ev = __expf(row[i] - m); row[i] = ev; ssum += ev; }
    float inv = 1.0f / ssum;
    for (int i = 0; i < 16; ++i) row[i] *= inv;
  }
  __syncthreads();
  // attn output o = a @ v
  for (int e = threadIdx.x; e < 16 * 128; e += 256) {
    int q = e >> 7, c = e & 127, hd = c >> 5;
    const float* a = sc + (hd * 16 + q) * 16;
    float s = 0.0f;
    for (int kk = 0; kk < 16; ++kk) s += a[kk] * qkv[kk * 384 + 256 + c];
    oo[e] = s; oh[e] = (_Float16)s;
  }
  __syncthreads();

  // 4) o @ Wo^T + bo + residual -> oo ; LN(g1,b1) -> xs/xh
  {
    int nt = wave;                              // 8 tiles, one per wave
    v8f acc = {};
    const _Float16* b0p = Wo + (size_t)nt * 16 * 128;
    for (int k = 0; k < 128; k += 32)
      acc = wmma16(load_frag_rm(oh + k, 128), load_frag_rm(b0p + k, 128), acc);
    int n = nt * 16 + (lane & 15); int mb = (lane & 16) ? 8 : 0;
    float bv = bo[n];
#pragma unroll
    for (int r = 0; r < 8; ++r) {
      int idx = (mb + r) * 128 + n;
      oo[idx] = xs[idx] + acc[r] + bv;
    }
  }
  __syncthreads();
  ln_rows16<128>(oo, g1, b1, xs, xh);
  __syncthreads();

  // 5) FF1: N=2048 -> 128 tiles, 16 per wave, ReLU, store f16 in LDS
  for (int nt = wave; nt < 128; nt += 8) {
    if (nt + 8 < 128) __builtin_prefetch(W1 + (size_t)(nt + 8) * 16 * 128, 0, 3);
    v8f acc = {};
    const _Float16* b0p = W1 + (size_t)nt * 16 * 128;
    for (int k = 0; k < 128; k += 32)
      acc = wmma16(load_frag_rm(xh + k, 128), load_frag_rm(b0p + k, 128), acc);
    int n = nt * 16 + (lane & 15); int mb = (lane & 16) ? 8 : 0;
    float bv = bf1[n];
#pragma unroll
    for (int r = 0; r < 8; ++r)
      hb[(mb + r) * 2048 + n] = (_Float16)fmaxf(acc[r] + bv, 0.0f);
  }
  __syncthreads();

  // 6) FF2: N=128 -> 8 tiles, K=2048, + residual -> oo ; LN(g2,b2) -> global x
  {
    int nt = wave;
    v8f acc = {};
    const _Float16* b0p = W2 + (size_t)nt * 16 * 2048;
    for (int k = 0; k < 2048; k += 32)
      acc = wmma16(load_frag_rm(hb + k, 2048), load_frag_rm(b0p + k, 2048), acc);
    int n = nt * 16 + (lane & 15); int mb = (lane & 16) ? 8 : 0;
    float bv = bf2[n];
#pragma unroll
    for (int r = 0; r < 8; ++r) {
      int idx = (mb + r) * 128 + n;
      oo[idx] = xs[idx] + acc[r] + bv;
    }
  }
  __syncthreads();
  ln_rows16<128>(oo, g2, b2, xg, nullptr);
}

// ---------------------------------------------------------------------------
// CLS head: 16 hops per block; h = x[:,0,:]; relu(h@W1^T+b1); LN; @W2^T+b2
// ---------------------------------------------------------------------------
__global__ __launch_bounds__(256)
void cls_head_kernel(const float* __restrict__ xbuf,
                     const _Float16* __restrict__ W1h, const float* __restrict__ b1,
                     const float* __restrict__ g, const float* __restrict__ b,
                     const _Float16* __restrict__ W2h, const float* __restrict__ b2,
                     float* __restrict__ out_tt)
{
  __shared__ __align__(16) _Float16 ah [16 * 128];
  __shared__ __align__(16) float    h2 [16 * 512];
  __shared__ __align__(16) _Float16 h2h[16 * 512];
  const int h0 = blockIdx.x * 16;
  const int wave = threadIdx.x >> 5, lane = threadIdx.x & 31;

  for (int i = threadIdx.x; i < 16 * 128; i += 256) {
    int r = i >> 7, c = i & 127;
    ah[i] = (_Float16)xbuf[(size_t)(h0 + r) * 16 * 128 + c];   // token 0
  }
  __syncthreads();
  for (int nt = wave; nt < 32; nt += 8) {                      // N=512
    v8f acc = {};
    const _Float16* b0p = W1h + (size_t)nt * 16 * 128;
    for (int k = 0; k < 128; k += 32)
      acc = wmma16(load_frag_rm(ah + k, 128), load_frag_rm(b0p + k, 128), acc);
    int n = nt * 16 + (lane & 15); int mb = (lane & 16) ? 8 : 0;
    float bv = b1[n];
#pragma unroll
    for (int r = 0; r < 8; ++r) h2[(mb + r) * 512 + n] = fmaxf(acc[r] + bv, 0.0f);
  }
  __syncthreads();
  ln_rows16<512>(h2, g, b, nullptr, h2h);
  __syncthreads();
  if (wave < 4) {                                              // N=64 -> 4 tiles
    int nt = wave;
    v8f acc = {};
    const _Float16* b0p = W2h + (size_t)nt * 16 * 512;
    for (int k = 0; k < 512; k += 32)
      acc = wmma16(load_frag_rm(h2h + k, 512), load_frag_rm(b0p + k, 512), acc);
    int n = nt * 16 + (lane & 15); int mb = (lane & 16) ? 8 : 0;
    float bv = b2[n];
#pragma unroll
    for (int r = 0; r < 8; ++r)
      out_tt[(size_t)(h0 + mb + r) * 64 + n] = acc[r] + bv;
  }
}

// ---------------------------------------------------------------------------
// Host launcher
// ---------------------------------------------------------------------------
static inline void* wsal(char*& p, size_t bytes) {
  void* r = (void*)p;
  p += (bytes + 255) & ~(size_t)255;
  return r;
}

extern "C" void kernel_launch(void* const* d_in, const int* in_sizes, int n_in,
                              void* d_out, int out_size, void* d_ws, size_t ws_size,
                              hipStream_t stream)
{
  (void)in_sizes; (void)n_in; (void)out_size; (void)ws_size;
  // ---- input pointers (insertion-order flattening of setup_inputs dict) ----
  const float* hf     = (const float*)d_in[0];
  const float* hs     = (const float*)d_in[1];
  const int*   hop_pi = (const int*)d_in[2];
  const int*   hop_st = (const int*)d_in[3];
  const int*   hop_po = (const int*)d_in[4];
  int pi = 5;
  const float* p_cls  = (const float*)d_in[pi++];
  const float* p_dc   = (const float*)d_in[pi++];
  const float* p_zero = (const float*)d_in[pi++];
  const float* p_one  = (const float*)d_in[pi++];
  const float* p_pos  = (const float*)d_in[pi++];
  const float* mW0  = (const float*)d_in[pi++]; const float* mb0  = (const float*)d_in[pi++];
  const float* mg0  = (const float*)d_in[pi++]; const float* mbn0 = (const float*)d_in[pi++];
  const float* mW1  = (const float*)d_in[pi++]; const float* mb1  = (const float*)d_in[pi++];
  const float* mg1  = (const float*)d_in[pi++]; const float* mbn1 = (const float*)d_in[pi++];
  const float* mW2  = (const float*)d_in[pi++]; const float* mb2  = (const float*)d_in[pi++];
  const float* eWqkv[3], *ebqkv[3], *eWo[3], *ebo[3], *eg1[3], *eb1[3];
  const float* eW1[3], *ebf1[3], *eW2[3], *ebf2[3], *eg2[3], *eb2[3];
  for (int l = 0; l < 3; ++l) {
    eWqkv[l] = (const float*)d_in[pi++]; ebqkv[l] = (const float*)d_in[pi++];
    eWo[l]   = (const float*)d_in[pi++]; ebo[l]   = (const float*)d_in[pi++];
    eg1[l]   = (const float*)d_in[pi++]; eb1[l]   = (const float*)d_in[pi++];
    eW1[l]   = (const float*)d_in[pi++]; ebf1[l]  = (const float*)d_in[pi++];
    eW2[l]   = (const float*)d_in[pi++]; ebf2[l]  = (const float*)d_in[pi++];
    eg2[l]   = (const float*)d_in[pi++]; eb2[l]   = (const float*)d_in[pi++];
  }
  const float* cW1 = (const float*)d_in[pi++]; const float* cb1 = (const float*)d_in[pi++];
  const float* cg  = (const float*)d_in[pi++]; const float* cb  = (const float*)d_in[pi++];
  const float* cW2 = (const float*)d_in[pi++]; const float* cb2 = (const float*)d_in[pi++];

  // ---- output layout: hs | hf | prob | hop_tt ----
  float* out  = (float*)d_out;
  float* o_hs   = out;
  float* o_hf   = out + (size_t)NN * DD;
  float* o_prob = out + (size_t)2 * NN * DD;
  float* o_tt   = o_prob + NN;

  // ---- workspace ----
  char* w = (char*)d_ws;
  float*    xbuf = (float*)   wsal(w, (size_t)HH * 16 * DD * 4);
  _Float16* hfb  = (_Float16*)wsal(w, (size_t)NN * DD * 2);
  float*    G    = (float*)   wsal(w, (size_t)NN * MLPH * 4);
  _Float16* Hh   = (_Float16*)wsal(w, (size_t)NN * MLPH * 2);
  _Float16* W0T  = (_Float16*)wsal(w, (size_t)MLPH * DD * 2);
  _Float16* W1T  = (_Float16*)wsal(w, (size_t)MLPH * MLPH * 2);
  _Float16 *hWqkv[3], *hWo[3], *hW1[3], *hW2[3];
  for (int l = 0; l < 3; ++l) {
    hWqkv[l] = (_Float16*)wsal(w, (size_t)3 * DD * DD * 2);
    hWo[l]   = (_Float16*)wsal(w, (size_t)DD * DD * 2);
    hW1[l]   = (_Float16*)wsal(w, (size_t)FFH * DD * 2);
    hW2[l]   = (_Float16*)wsal(w, (size_t)DD * FFH * 2);
  }
  _Float16* hCW1 = (_Float16*)wsal(w, (size_t)4 * DD * DD * 2);
  _Float16* hCW2 = (_Float16*)wsal(w, (size_t)TTN * 4 * DD * 2);
  int* kvmask    = (int*)     wsal(w, (size_t)HH * 4);

  // ---- passthrough outputs ----
  hipMemcpyAsync(o_hs, hs, (size_t)NN * DD * 4, hipMemcpyDeviceToDevice, stream);
  hipMemcpyAsync(o_hf, hf, (size_t)NN * DD * 4, hipMemcpyDeviceToDevice, stream);

  // ---- weight / activation conversion ----
  {
    long n = (long)NN * DD;
    conv_f16_kernel<<<(int)((n + 255) / 256), 256, 0, stream>>>(hf, hfb, n);
  }
  tconv_f16_kernel<<<(DD * MLPH + 255) / 256, 256, 0, stream>>>(mW0, W0T, DD, MLPH);
  tconv_f16_kernel<<<(MLPH * MLPH + 255) / 256, 256, 0, stream>>>(mW1, W1T, MLPH, MLPH);
  for (int l = 0; l < 3; ++l) {
    conv_f16_kernel<<<(3 * DD * DD + 255) / 256, 256, 0, stream>>>(eWqkv[l], hWqkv[l], 3 * DD * DD);
    conv_f16_kernel<<<(DD * DD + 255) / 256, 256, 0, stream>>>(eWo[l], hWo[l], DD * DD);
    conv_f16_kernel<<<(FFH * DD + 255) / 256, 256, 0, stream>>>(eW1[l], hW1[l], (long)FFH * DD);
    conv_f16_kernel<<<(DD * FFH + 255) / 256, 256, 0, stream>>>(eW2[l], hW2[l], (long)DD * FFH);
  }
  conv_f16_kernel<<<(4 * DD * DD + 255) / 256, 256, 0, stream>>>(cW1, hCW1, 4 * DD * DD);
  conv_f16_kernel<<<(TTN * 4 * DD + 255) / 256, 256, 0, stream>>>(cW2, hCW2, TTN * 4 * DD);

  // ---- mlp3: prob ----
  {
    const int total = (NN / 16) * (MLPH / 64);   // 50000 waves (16x64 strips)
    const int blks  = total / 8;                 // 6250
    gemm16x4_kernel<<<blks, 256, 0, stream>>>(hfb, W0T, mb0, G, NN, MLPH, DD);
    ln_relu_kernel<<<NN, 256, 0, stream>>>(G, mg0, mbn0, Hh);
    gemm16x4_kernel<<<blks, 256, 0, stream>>>(Hh, W1T, mb1, G, NN, MLPH, MLPH);
    ln_relu_kernel<<<NN, 256, 0, stream>>>(G, mg1, mbn1, Hh);
    prob_kernel<<<NN / 8, 256, 0, stream>>>(Hh, mW2, mb2, o_prob, NN);
  }

  // ---- hop encoder ----
  build_seq_kernel<<<HH, 128, 0, stream>>>(hf, hop_pi, hop_st, hop_po,
                                           p_cls, p_dc, p_zero, p_one, p_pos,
                                           xbuf, kvmask);
  const int dyn_lds = 16 * 128 * 4 + 16 * 2048 * 2;   // xs + hb = 72 KB
  for (int l = 0; l < 3; ++l) {
    enc_layer_kernel<<<HH, 256, dyn_lds, stream>>>(
        xbuf, kvmask,
        hWqkv[l], ebqkv[l], hWo[l], ebo[l], eg1[l], eb1[l],
        hW1[l], ebf1[l], hW2[l], ebf2[l], eg2[l], eb2[l]);
  }
  cls_head_kernel<<<HH / 16, 256, 0, stream>>>(xbuf, hCW1, cb1, cg, cb, hCW2, cb2, o_tt);
}
